// HGATActor_82566451298742
// MI455X (gfx1250) — matrix operand order repaired and checked by reading, not compiled, not run on hardware
//
#include <hip/hip_runtime.h>
#include <hip/hip_bf16.h>
#include <math.h>

// ---------------------------------------------------------------------------
// Problem constants (from reference): B=32, O=200, M=50, D=128
// ---------------------------------------------------------------------------
#define BB 32
#define OO 200
#define MM 50
#define DD 128

typedef __attribute__((ext_vector_type(16))) __bf16 v16bf;
typedef __attribute__((ext_vector_type(8)))  float  v8f;

union V16 {
    unsigned short u[16];
    v16bf v;
};

// Round-to-nearest-even f32 -> bf16 bits
__device__ __forceinline__ unsigned short bf16_rne(float x) {
    unsigned u = __builtin_bit_cast(unsigned, x);
    unsigned r = u + 0x7FFFu + ((u >> 16) & 1u);
    return (unsigned short)(r >> 16);
}
__device__ __forceinline__ float bf16_to_f(unsigned short h) {
    unsigned u = ((unsigned)h) << 16;
    return __builtin_bit_cast(float, u);
}

// ---------------------------------------------------------------------------
// K0: c[b,k] = b1[k] + sum_j g[b,j] * Wg[j,k]     (Wg = W1[256:512])
// grid = (B), block = (D)
// ---------------------------------------------------------------------------
__global__ void hgat_cvec(const float* __restrict__ g,
                          const float* __restrict__ W1,
                          const float* __restrict__ b1,
                          float* __restrict__ C) {
    const int b = blockIdx.x;
    const int k = threadIdx.x;
    const float* gr = g + (size_t)b * (2 * DD);
    const float* Wg = W1 + (size_t)(2 * DD) * DD;  // rows 256..511
    float s = b1[k];
#pragma unroll 8
    for (int j = 0; j < 2 * DD; ++j)
        s = fmaf(gr[j], Wg[(size_t)j * DD + k], s);
    C[(size_t)b * DD + k] = s;
}

// ---------------------------------------------------------------------------
// K1: split-bf16 WMMA GEMM.
//   blocks [0,400):  PO[row,:]  = h_o_flat(6400x128) @ Wo(=W1[0:128])  + C[b]
//   blocks [400,500):PM[row,:]  = h_m_flat(1600x128) @ Wm(=W1[128:256])
// block = 256 threads = 8 waves; wave w owns column tile n0 = 16*w.
// Each wave: 4 K-chunks of 32, 3 WMMAs per chunk (hi*hi + hi*lo + lo*hi).
// ---------------------------------------------------------------------------
__global__ void hgat_gemm(const float* __restrict__ h_o,
                          const float* __restrict__ h_m,
                          const float* __restrict__ W1,
                          const float* __restrict__ C,
                          float* __restrict__ PO,
                          float* __restrict__ PM) {
    const int blk  = blockIdx.x;
    const int tid  = threadIdx.x;
    const int lane = tid & 31;
    const int wave = tid >> 5;
    const int n0   = wave * 16;

    const bool isPO = (blk < 400);
    const float* X  = isPO ? h_o : h_m;
    const float* Wp = isPO ? W1 : (W1 + (size_t)DD * DD);  // Wo or Wm
    float*       Y  = isPO ? PO : PM;
    const int    r0 = (isPO ? blk : (blk - 400)) * 16;

    const int mrow = r0 + (lane & 15);
    const int ncol = n0 + (lane & 15);
    const bool hi_half = (lane >= 16);

    const float* arow = X + (size_t)mrow * DD;

    v8f acc = {0.f, 0.f, 0.f, 0.f, 0.f, 0.f, 0.f, 0.f};

#pragma unroll
    for (int kc = 0; kc < 4; ++kc) {
        const int kbase = kc * 32;

        // ---- A (16x32 bf16 layout): lane<16 -> K {0..7,16..23}; lane>=16 -> K {8..15,24..31}
        const int abase = kbase + (hi_half ? 8 : 0);
        float av[16];
        {
            float4 a0 = *(const float4*)(arow + abase);
            float4 a1 = *(const float4*)(arow + abase + 4);
            float4 a2 = *(const float4*)(arow + abase + 16);
            float4 a3 = *(const float4*)(arow + abase + 20);
            av[0] = a0.x;  av[1] = a0.y;  av[2]  = a0.z;  av[3]  = a0.w;
            av[4] = a1.x;  av[5] = a1.y;  av[6]  = a1.z;  av[7]  = a1.w;
            av[8] = a2.x;  av[9] = a2.y;  av[10] = a2.z;  av[11] = a2.w;
            av[12] = a3.x; av[13] = a3.y; av[14] = a3.z;  av[15] = a3.w;
        }
        V16 ahi, alo;
#pragma unroll
        for (int e = 0; e < 16; ++e) {
            unsigned short h = bf16_rne(av[e]);
            ahi.u[e] = h;
            alo.u[e] = bf16_rne(av[e] - bf16_to_f(h));
        }

        // ---- B (32x16 bf16 layout): lane<16 -> K {0..7,16..23}; lane>=16 -> K {8..15,24..31}; N = lane&15
        const int bbase = kbase + (hi_half ? 8 : 0);
        V16 bhi, blo;
#pragma unroll
        for (int e = 0; e < 16; ++e) {
            float w = Wp[(size_t)(bbase + (e < 8 ? e : e + 8)) * DD + ncol];
            unsigned short h = bf16_rne(w);
            bhi.u[e] = h;
            blo.u[e] = bf16_rne(w - bf16_to_f(h));
        }

        // ---- split-precision accumulation: A*B ~= Ahi*Bhi + Ahi*Blo + Alo*Bhi
        acc = __builtin_amdgcn_wmma_f32_16x16x32_bf16(false, ahi.v, false, bhi.v,
                                                      (short)0, acc, false, false);
        acc = __builtin_amdgcn_wmma_f32_16x16x32_bf16(false, ahi.v, false, blo.v,
                                                      (short)0, acc, false, false);
        acc = __builtin_amdgcn_wmma_f32_16x16x32_bf16(false, alo.v, false, bhi.v,
                                                      (short)0, acc, false, false);
    }

    // ---- store C/D tile: VGPR r = row (r | r+8), col = n0 + (lane&15)
    const int rbase = hi_half ? 8 : 0;
#pragma unroll
    for (int r = 0; r < 8; ++r) {
        const int row = r0 + rbase + r;
        float v = acc[r];
        if (isPO) {
            const int b = row / OO;            // batch of this h_o row
            v += C[(size_t)b * DD + ncol];     // + (g@Wg + b1) broadcast
            PO[(size_t)row * DD + ncol] = v;
        } else {
            PM[(size_t)row * DD + ncol] = v;
        }
    }
}

// ---------------------------------------------------------------------------
// K2: logits[b,o,m] = b2 + sum_k tanh(PO[b,o,k] + PM[b,m,k]) * W2[k], masked.
// grid = (B*O), block = 64; PO row + W2 staged in LDS, thread t handles m=t.
// ---------------------------------------------------------------------------
__global__ void hgat_logits(const float* __restrict__ PO,
                            const float* __restrict__ PM,
                            const float* __restrict__ W2,
                            const float* __restrict__ b2,
                            const unsigned char* __restrict__ legal,
                            float* __restrict__ LG) {
    __shared__ float po_s[DD];
    __shared__ float w2_s[DD];
    const int t = threadIdx.x;
    const int b = blockIdx.x / OO;
    const int o = blockIdx.x % OO;

    for (int i = t; i < DD; i += 64) {
        po_s[i] = PO[(size_t)blockIdx.x * DD + i];
        w2_s[i] = W2[i];
    }
    __syncthreads();

    if (t < MM) {
        const float* pm = PM + ((size_t)b * MM + t) * DD;
        float s = b2[0];
#pragma unroll 8
        for (int k = 0; k < DD; ++k)
            s = fmaf(tanhf(po_s[k] + pm[k]), w2_s[k], s);
        const size_t idx = (size_t)b * (OO * MM) + (size_t)o * MM + t;
        LG[idx] = legal[idx] ? s : -__builtin_inff();
    }
}

// ---------------------------------------------------------------------------
// K3: per-batch masked softmax over O*M = 10000 logits.
// grid = (B), block = 256. exp values cached back into LG.
// ---------------------------------------------------------------------------
__global__ void hgat_softmax(float* __restrict__ LG, float* __restrict__ out) {
    __shared__ float red[256];
    const int t = threadIdx.x;
    const int n = OO * MM;
    float* L = LG + (size_t)blockIdx.x * n;
    float* P = out + (size_t)blockIdx.x * n;

    // pass 1: max
    float mx = -__builtin_inff();
    for (int i = t; i < n; i += 256) mx = fmaxf(mx, L[i]);
    red[t] = mx;
    __syncthreads();
    for (int s = 128; s > 0; s >>= 1) {
        if (t < s) red[t] = fmaxf(red[t], red[t + s]);
        __syncthreads();
    }
    mx = red[0];
    __syncthreads();

    // pass 2: sum of exp (cache exp in LG)
    float sm = 0.f;
    for (int i = t; i < n; i += 256) {
        float e = expf(L[i] - mx);
        L[i] = e;
        sm += e;
    }
    red[t] = sm;
    __syncthreads();
    for (int s = 128; s > 0; s >>= 1) {
        if (t < s) red[t] += red[t + s];
        __syncthreads();
    }
    const float inv = 1.f / red[0];
    __syncthreads();

    // pass 3: normalize
    for (int i = t; i < n; i += 256) P[i] = L[i] * inv;
}

// ---------------------------------------------------------------------------
// launch
// ---------------------------------------------------------------------------
extern "C" void kernel_launch(void* const* d_in, const int* in_sizes, int n_in,
                              void* d_out, int out_size, void* d_ws, size_t ws_size,
                              hipStream_t stream) {
    const float*         h_o   = (const float*)d_in[0];          // (B,O,D)
    const float*         h_m   = (const float*)d_in[1];          // (B,M,D)
    const float*         g     = (const float*)d_in[2];          // (B,2D)
    const unsigned char* legal = (const unsigned char*)d_in[3];  // (B,O,M) bool
    const float*         W1    = (const float*)d_in[4];          // (4D,D)
    const float*         b1    = (const float*)d_in[5];          // (D,)
    const float*         W2    = (const float*)d_in[6];          // (D,1)
    const float*         b2    = (const float*)d_in[7];          // (1,)
    float*               out   = (float*)d_out;                  // (B, O*M)

    // workspace layout (floats): C | PO | PM | LG  (~5.4 MB total)
    float* ws = (float*)d_ws;
    float* C  = ws;                                // BB*DD          = 4096
    float* PO = C + (size_t)BB * DD;               // BB*OO*DD       = 819200
    float* PM = PO + (size_t)BB * OO * DD;         // BB*MM*DD       = 204800
    float* LG = PM + (size_t)BB * MM * DD;         // BB*OO*MM       = 320000

    hgat_cvec<<<BB, DD, 0, stream>>>(g, W1, b1, C);
    hgat_gemm<<<(BB * OO / 16) + (BB * MM / 16), 256, 0, stream>>>(h_o, h_m, W1, C, PO, PM);
    hgat_logits<<<BB * OO, 64, 0, stream>>>(PO, PM, W2, b2, legal, LG);
    hgat_softmax<<<BB, 256, 0, stream>>>(LG, out);
}
